// Joiner_11940009083043
// MI455X (gfx1250) — compile-verified, matrix operand order checked by reading, not run on hardware
//
#include <hip/hip_runtime.h>

typedef __attribute__((ext_vector_type(16))) _Float16 v16h;
typedef __attribute__((ext_vector_type(8)))  float    v8f;

#define B_   4
#define T_   400
#define U_   100
#define D_   512
#define V_   1024
#define ROWS (B_ * T_ * U_)        // 160000
#define KB   (D_ / 32)             // 16 k-blocks of 32
#define NB   (V_ / 16)             // 64 column blocks of 16

// CDNA5 hardware tanh (TRANS32 op). The trailing v_nop inside the asm
// guarantees the required >=1 instruction gap between a TRANS op and the
// first consumer of its result (co-execution hazard rule, ISA ch.5 §7.4.3).
__device__ __forceinline__ float htanh(float x) {
  float r;
  asm("v_tanh_f32 %0, %1\n\tv_nop" : "=v"(r) : "v"(x));
  return r;
}

// ---------------------------------------------------------------------------
// Pre-pass: convert W (V x D, f32, row-major) into f16, swizzled so that the
// GEMM kernel's B-matrix fragment for (kb, nb) is 32 lanes x 16 f16 contiguous.
// Per ISA 7.12.2 (16-bit 32x16 B): lane l holds column v = nb*16 + (l&15);
// element i holds K = kb*32 + 8*(l>>4) + ((i<8) ? i : 16 + (i-8)).
// ---------------------------------------------------------------------------
__global__ __launch_bounds__(256) void swizzle_w_kernel(
    const float* __restrict__ W, _Float16* __restrict__ wsw) {
  int id   = blockIdx.x * 256 + threadIdx.x;  // 0..32767
  int lane = id & 31;
  int frag = id >> 5;                          // 0..1023 = kb*NB + nb
  int kb   = frag >> 6;
  int nb   = frag & (NB - 1);
  int v    = nb * 16 + (lane & 15);
  int half = lane >> 4;
  int base0 = kb * 32 + 8 * half;
  const float* wrow = W + (size_t)v * D_;
  v16h f;
#pragma unroll
  for (int i = 0; i < 8; ++i) f[i]     = (_Float16)wrow[base0 + i];
#pragma unroll
  for (int i = 0; i < 8; ++i) f[8 + i] = (_Float16)wrow[base0 + 16 + i];
  ((v16h*)wsw)[frag * 32 + lane] = f;
}

// ---------------------------------------------------------------------------
// Main kernel: block = 32 rows x 512 cols of output. 8 waves; each wave owns
// 2 M-tiles x 4 N-tiles (16x16 each), accumulating over K=512 with
// v_wmma_f32_16x16x32_f16. A tile (tanh(src+tgt) in f16) staged in LDS in
// WMMA fragment order (same per-lane K mapping as B).
// ---------------------------------------------------------------------------
__global__ __launch_bounds__(256) void joiner_gemm_kernel(
    const float* __restrict__ src,   // (B,T,D)
    const float* __restrict__ tgt,   // (B,U,D)
    const _Float16* __restrict__ wsw,// swizzled W, f16
    const float* __restrict__ bias,  // (V,)
    float* __restrict__ out) {       // (B,T,U,V)
  __shared__ v16h ldsA[2 * KB * 32];            // 2 tiles * 16 kb * 32 lanes * 32B = 32 KB
  const int tid  = threadIdx.x;
  const int row0 = blockIdx.x * 32;

  // ---- Stage A tile: each thread builds 4 fragments (16 f16 each) ----
#pragma unroll
  for (int j = 0; j < 4; ++j) {
    int f    = tid + 256 * j;                   // f = tile*512 + kb*32 + lane
    int lane = f & 31;
    int kb   = (f >> 5) & (KB - 1);
    int m    = row0 + (f >> 9) * 16 + (lane & 15);
    int b    = m / (T_ * U_);
    int rem  = m % (T_ * U_);
    int t    = rem / U_;
    int u    = rem % U_;
    const float* sp = src + (size_t)(b * T_ + t) * D_;
    const float* tp = tgt + (size_t)(b * U_ + u) * D_;
    int base0 = kb * 32 + 8 * (lane >> 4);
    v16h av;
#pragma unroll
    for (int i = 0; i < 8; ++i)
      av[i]     = (_Float16)htanh(sp[base0 + i]      + tp[base0 + i]);
#pragma unroll
    for (int i = 0; i < 8; ++i)
      av[8 + i] = (_Float16)htanh(sp[base0 + 16 + i] + tp[base0 + 16 + i]);
    ldsA[f] = av;
  }
  __syncthreads();

  // ---- WMMA main loop ----
  const int lane   = tid & 31;
  const int wave   = tid >> 5;
  const int nbBase = blockIdx.y * 32 + wave * 4;  // 4 N-tiles per wave

  v8f acc[2][4] = {};
  for (int kb = 0; kb < KB; ++kb) {
    v16h a0 = ldsA[(0 * KB + kb) * 32 + lane];
    v16h a1 = ldsA[(1 * KB + kb) * 32 + lane];
#pragma unroll
    for (int j = 0; j < 4; ++j) {
      v16h bf = ((const v16h*)wsw)[(kb * NB + nbBase + j) * 32 + lane];
      acc[0][j] = __builtin_amdgcn_wmma_f32_16x16x32_f16(
          false, a0, false, bf, (short)0, acc[0][j], false, false);
      acc[1][j] = __builtin_amdgcn_wmma_f32_16x16x32_f16(
          false, a1, false, bf, (short)0, acc[1][j], false, false);
    }
  }

  // ---- Epilogue: bias + nontemporal store (write-once output, keep W in L2)
  // C/D layout: lanes 0-15 hold rows 0-7 (VGPR r), lanes 16-31 rows 8-15.
  const int half = lane >> 4;
#pragma unroll
  for (int j = 0; j < 4; ++j) {
    int v = (nbBase + j) * 16 + (lane & 15);
    float bv = bias[v];
#pragma unroll
    for (int tile = 0; tile < 2; ++tile) {
#pragma unroll
      for (int r = 0; r < 8; ++r) {
        int m = row0 + tile * 16 + half * 8 + r;
        __builtin_nontemporal_store(acc[tile][j][r] + bv,
                                    out + (size_t)m * V_ + v);
      }
    }
  }
}

// Copy the pass-through length vectors into the tail of d_out.
__global__ void copy_lengths_kernel(const int* __restrict__ sl,
                                    const int* __restrict__ tl,
                                    int* __restrict__ out_tail) {
  int i = threadIdx.x;
  if (i < B_)            out_tail[i] = sl[i];
  else if (i < 2 * B_)   out_tail[i] = tl[i - B_];
}

extern "C" void kernel_launch(void* const* d_in, const int* in_sizes, int n_in,
                              void* d_out, int out_size, void* d_ws, size_t ws_size,
                              hipStream_t stream) {
  const float* src  = (const float*)d_in[0];   // (B,T,D) f32
  const int*   slen = (const int*)  d_in[1];   // (B,)
  const float* tgt  = (const float*)d_in[2];   // (B,U,D) f32
  const int*   tlen = (const int*)  d_in[3];   // (B,)
  const float* W    = (const float*)d_in[4];   // (V,D) f32
  const float* bias = (const float*)d_in[5];   // (V,)

  float*     out = (float*)d_out;
  _Float16*  wsw = (_Float16*)d_ws;            // 1 MB swizzled f16 W

  // 1) Convert + swizzle W into workspace (1024 frags * 32 lanes = 32768 thr).
  swizzle_w_kernel<<<128, 256, 0, stream>>>(W, wsw);

  // 2) Main fused tanh-join + GEMM. 160000/32 = 5000 row tiles, 2 col halves.
  joiner_gemm_kernel<<<dim3(ROWS / 32, 2), 256, 0, stream>>>(
      src, tgt, wsw, bias, out);

  // 3) Pass-through lengths at the tail of d_out.
  copy_lengths_kernel<<<1, 2 * B_, 0, stream>>>(
      slen, tlen, (int*)(out + (size_t)ROWS * V_));
}
